// ada_d_conv_layer_50706383897208
// MI455X (gfx1250) — matrix operand order repaired and checked by reading, not compiled, not run on hardware
//
#include <hip/hip_runtime.h>

// ---------------------------------------------------------------------------
// Problem constants (reference: N=100000, E=1600000, IN_F=256, HID=128)
// ---------------------------------------------------------------------------
#define IN_F 256
#define HID  128
#define XROW (2 * IN_F)      // 512 floats per x row
#define LDSW 264             // 256 + 8 bf16 pad -> row stride 528B = 132 dwords (banks spread)

typedef __attribute__((ext_vector_type(16))) __bf16 v16bf;
typedef __attribute__((ext_vector_type(8)))  __bf16 v8bf;
typedef __attribute__((ext_vector_type(8)))  float  v8f;

// ---------------------------------------------------------------------------
// Pack W [256,128] fp32 (row-major, K x N) into per-lane WMMA B-operand order:
//   index t = ((ntile*8 + kstep)*32 + lane), 16 bf16 per t.
//   B layout: lane<16 -> col = ntile*16 + lane, K = kstep*32 + j (j=0..15)
//             lane>=16 -> same col (lane&15),  K = kstep*32 + 16 + j
// ---------------------------------------------------------------------------
__global__ __launch_bounds__(256)
void gcn_pack_w_kernel(const float* __restrict__ W, __bf16* __restrict__ Wp) {
    int t = blockIdx.x * 256 + threadIdx.x;       // 0 .. 2047
    if (t >= 8 * 8 * 32) return;
    const int lane  = t & 31;
    const int ks    = (t >> 5) & 7;
    const int ntile = t >> 8;
    const int kbase = ks * 32 + ((lane & 16) ? 16 : 0);
    const int col   = ntile * 16 + (lane & 15);
    __bf16* dst = Wp + (size_t)t * 16;
#pragma unroll
    for (int j = 0; j < 16; ++j)
        dst[j] = (__bf16)W[(size_t)(kbase + j) * HID + col];
}

// ---------------------------------------------------------------------------
// h = x[:, col_base:col_base+256] @ W + bias   via v_wmma_f32_16x16x32_bf16
// Block = 256 threads = 8 waves. Each block: one 16-row strip of x staged to
// LDS as bf16 (converted once, reused by all 8 waves). Wave w computes the
// 16x16 output tile for columns [16w, 16w+16).  K loop: 8 x WMMA (K=32 each).
// ---------------------------------------------------------------------------
__global__ __launch_bounds__(256)
void gcn_gemm_bias_kernel(const float* __restrict__ x,
                          const __bf16* __restrict__ Wp,
                          const float* __restrict__ bias,
                          float* __restrict__ h,
                          int n_rows, int col_base) {
    __shared__ __bf16 lA[16 * LDSW];

    const int tid = threadIdx.x;
    const int m0  = blockIdx.x * 16;

    // Cooperative load: 16 rows x 256 cols fp32 -> bf16 in LDS (16 elems/thread)
    {
        const int row = tid >> 4;              // 0..15
        const int cs  = (tid & 15) * 16;       // 0,16,...,240
        int gr = m0 + row;
        if (gr >= n_rows) gr = n_rows - 1;     // clamp (N is a multiple of 16 anyway)
        const float* src = x + (size_t)gr * XROW + col_base + cs;
        __bf16* dst = &lA[row * LDSW + cs];
#pragma unroll
        for (int j = 0; j < 16; ++j) dst[j] = (__bf16)src[j];
    }
    __syncthreads();

    const int lane  = tid & 31;
    const int ntile = tid >> 5;                // wave id = output column tile
    const int lh    = lane >> 4;               // lane half
    const int mr    = lane & 15;

    // Packed B base: vector index (ntile*8 + ks)*32 + lane
    const v16bf* bp = (const v16bf*)Wp + (size_t)ntile * 8 * 32 + lane;

    v8f c = {};
#pragma unroll
    for (int ks = 0; ks < 8; ++ks) {
        // A operand, 16x32 bf16 layout:
        //   lane<16: row mr, K = ks*32 + {0..15}
        //   lane>=16: row mr, K = ks*32 + {16..31}
        const v8bf a0 = *(const v8bf*)&lA[mr * LDSW + ks * 32 + lh * 16];
        const v8bf a1 = *(const v8bf*)&lA[mr * LDSW + ks * 32 + lh * 16 + 8];
        v16bf a;
#pragma unroll
        for (int j = 0; j < 8; ++j) { a[j] = a0[j]; a[8 + j] = a1[j]; }

        const v16bf b = bp[ks * 32];

        c = __builtin_amdgcn_wmma_f32_16x16x32_bf16(
                /*neg_a=*/false, a, /*neg_b=*/false, b,
                /*c_mod=*/(short)0, c, /*reuse_a=*/false, /*reuse_b=*/false);
    }

    // Epilogue: C/D layout -> vgpr i, lane L: M = lh*8 + i, N = L%16. Add bias.
    const int   col = ntile * 16 + mr;
    const float bv  = bias[col];
#pragma unroll
    for (int i = 0; i < 8; ++i) {
        const int r = m0 + lh * 8 + i;
        if (r < n_rows) h[(size_t)r * HID + col] = c[i] + bv;
    }
}

// ---------------------------------------------------------------------------
// Fused SpMM scatter for both adjacencies:
//   out[row] += val * h[col]   (128 floats per edge)
// One wave per edge; lane handles 4 contiguous floats (float4 gather from h,
// 4x global_atomic_add_f32 scatter). h1/h2 (102MB) + out (51MB) live in the
// 192MB L2, so the random traffic stays on-chip.
// ---------------------------------------------------------------------------
__global__ __launch_bounds__(256)
void gcn_spmm_scatter_kernel(const int* __restrict__ rows1, const int* __restrict__ cols1,
                             const float* __restrict__ vals1,
                             const int* __restrict__ rows2, const int* __restrict__ cols2,
                             const float* __restrict__ vals2,
                             const float* __restrict__ h1, const float* __restrict__ h2,
                             float* __restrict__ out, int E) {
    const int  lane = threadIdx.x & 31;
    long long  w    = (long long)blockIdx.x * (blockDim.x >> 5) + (threadIdx.x >> 5);
    const long long nw    = (long long)gridDim.x * (blockDim.x >> 5);
    const long long total = 2LL * E;

    for (long long e = w; e < total; e += nw) {
        const int* rows; const int* cols; const float* vals; const float* h;
        long long  i;
        if (e < E) { rows = rows1; cols = cols1; vals = vals1; h = h1; i = e;     }
        else       { rows = rows2; cols = cols2; vals = vals2; h = h2; i = e - E; }

        const int   r  = rows[i];
        const int   cI = cols[i];
        const float v  = vals[i];

        const float4 hv = *(const float4*)(h + (size_t)cI * HID + lane * 4);
        float* op = out + (size_t)r * HID + lane * 4;
        unsafeAtomicAdd(op + 0, hv.x * v);
        unsafeAtomicAdd(op + 1, hv.y * v);
        unsafeAtomicAdd(op + 2, hv.z * v);
        unsafeAtomicAdd(op + 3, hv.w * v);
    }
}

// ---------------------------------------------------------------------------
// kernel_launch
// Inputs: 0:x 1:W1 2:b1 3:W2 4:b2 5:adj1_rows 6:adj1_cols 7:adj1_vals
//         8:adj2_rows 9:adj2_cols 10:adj2_vals
// Workspace: h1 [N*128 f32] | h2 [N*128 f32] | Wp1 [64KB bf16] | Wp2 [64KB bf16]
// ---------------------------------------------------------------------------
extern "C" void kernel_launch(void* const* d_in, const int* in_sizes, int n_in,
                              void* d_out, int out_size, void* d_ws, size_t ws_size,
                              hipStream_t stream) {
    const float* x  = (const float*)d_in[0];
    const float* W1 = (const float*)d_in[1];
    const float* b1 = (const float*)d_in[2];
    const float* W2 = (const float*)d_in[3];
    const float* b2 = (const float*)d_in[4];
    const int*   r1 = (const int*)d_in[5];
    const int*   c1 = (const int*)d_in[6];
    const float* v1 = (const float*)d_in[7];
    const int*   r2 = (const int*)d_in[8];
    const int*   c2 = (const int*)d_in[9];
    const float* v2 = (const float*)d_in[10];
    float* out = (float*)d_out;

    const int E = in_sizes[5];
    const int N = out_size / HID;

    char*   ws  = (char*)d_ws;
    float*  h1  = (float*)ws;
    float*  h2  = h1 + (size_t)N * HID;
    __bf16* Wp1 = (__bf16*)(ws + 2 * (size_t)N * HID * sizeof(float));
    __bf16* Wp2 = Wp1 + 8 * 8 * 32 * 16;

    // out = 0 (harness poisons it; also gives us the scatter accumulation base)
    hipMemsetAsync(d_out, 0, (size_t)out_size * sizeof(float), stream);

    // Pack weights into WMMA B-operand bf16 layout (128KB total, once per call)
    gcn_pack_w_kernel<<<8, 256, 0, stream>>>(W1, Wp1);
    gcn_pack_w_kernel<<<8, 256, 0, stream>>>(W2, Wp2);

    // Dense GEMMs + bias -> h1, h2 (bf16 WMMA, fp32 accumulate)
    const int mtiles = (N + 15) / 16;
    gcn_gemm_bias_kernel<<<mtiles, 256, 0, stream>>>(x, Wp1, b1, h1, N, 0);
    gcn_gemm_bias_kernel<<<mtiles, 256, 0, stream>>>(x, Wp2, b2, h2, N, IN_F);

    // Fused dual-adjacency SpMM scatter into out
    gcn_spmm_scatter_kernel<<<8192, 256, 0, stream>>>(r1, c1, v1, r2, c2, v2,
                                                      h1, h2, out, E);
}
